// Head_76991583748457
// MI455X (gfx1250) — compile-verified, matrix operand order checked by reading, not run on hardware
//
#include <hip/hip_runtime.h>

#define Bb  8
#define Tt  4096
#define Cc  1024
#define HSs 64
#define KQ  2048   // ceil(T/2)

typedef __attribute__((ext_vector_type(16))) __bf16 v16bf;
typedef __attribute__((ext_vector_type(8)))  __bf16 v8bf;
typedef __attribute__((ext_vector_type(2)))  __bf16 v2bf;
typedef __attribute__((ext_vector_type(8)))  float  v8f;
typedef __attribute__((ext_vector_type(4)))  int    v4i;

typedef __attribute__((address_space(1))) v4i gv4i;   // global 128b chunk
typedef __attribute__((address_space(3))) v4i sv4i;   // LDS 128b chunk

#ifndef __has_builtin
#define __has_builtin(x) 0
#endif

#if __has_builtin(__builtin_amdgcn_cvt_pk_bf16_f32)
#define HAVE_PK 1
#else
#define HAVE_PK 0
#endif

#if __has_builtin(__builtin_amdgcn_global_load_async_to_lds_b128)
#define HAVE_ASYNC 1
#else
#define HAVE_ASYNC 0
#endif

#if __has_builtin(__builtin_amdgcn_s_wait_asynccnt)
#define WAIT_ASYNC(n) __builtin_amdgcn_s_wait_asynccnt(n)
#else
#define WAIT_ASYNC(n) asm volatile("s_wait_asynccnt " #n ::: "memory")
#endif

// ---- bf16 conversion helpers --------------------------------------------

__device__ __forceinline__ v2bf pk2(float a, float b) {
#if HAVE_PK
  return __builtin_amdgcn_cvt_pk_bf16_f32(a, b);
#else
  union { float f; unsigned u; } x, y; x.f = a; y.f = b;
  unsigned ra = (x.u + 0x7FFFu + ((x.u >> 16) & 1u)) >> 16;
  unsigned rb = (y.u + 0x7FFFu + ((y.u >> 16) & 1u)) >> 16;
  union { unsigned u; v2bf v; } o; o.u = (ra & 0xFFFFu) | (rb << 16);
  return o.v;
#endif
}

__device__ __forceinline__ __bf16 f2bf(float f) {
  v2bf t = pk2(f, f);
  return t[0];
}

__device__ __forceinline__ v16bf pack16(v8bf lo, v8bf hi) {
  v16bf r;
#pragma unroll
  for (int i = 0; i < 8; ++i) { r[i] = lo[i]; r[i + 8] = hi[i]; }
  return r;
}

// cross-lane reduction within each 16-lane half (wave32), via ds_swizzle
#define XF(v, imm) __int_as_float(__builtin_amdgcn_ds_swizzle(__float_as_int(v), (imm)))

__device__ __forceinline__ float rmax16(float v) {
  v = fmaxf(v, XF(v, 0x041F));
  v = fmaxf(v, XF(v, 0x081F));
  v = fmaxf(v, XF(v, 0x101F));
  v = fmaxf(v, XF(v, 0x201F));
  return v;
}
__device__ __forceinline__ float rsum16(float v) {
  v += XF(v, 0x041F);
  v += XF(v, 0x081F);
  v += XF(v, 0x101F);
  v += XF(v, 0x201F);
  return v;
}

// ---- kernel: f32 -> bf16 streaming convert ------------------------------

__global__ void cvt_kernel(const float* __restrict__ src, __bf16* __restrict__ dst,
                           long long n) {
  long long i = ((long long)blockIdx.x * blockDim.x + threadIdx.x) * 8;
  if (i >= n) return;
  float4 u = *(const float4*)(src + i);
  float4 v = *(const float4*)(src + i + 4);
  v2bf p0 = pk2(u.x, u.y), p1 = pk2(u.z, u.w);
  v2bf p2 = pk2(v.x, v.y), p3 = pk2(v.z, v.w);
  v8bf o;
  o[0] = p0[0]; o[1] = p0[1]; o[2] = p1[0]; o[3] = p1[1];
  o[4] = p2[0]; o[5] = p2[1]; o[6] = p3[0]; o[7] = p3[1];
  *(v8bf*)(dst + i) = o;
}

// ---- kernel: keep indices -----------------------------------------------

__global__ void keep_kernel(int* __restrict__ keep) {
  int i = blockIdx.x * blockDim.x + threadIdx.x;
  if (i >= KQ) return;
  const int a = (Tt + 3) / 4;
  int x = KQ - 1 - i;                 // reversed order
  int v;
  if (x < a) {
    v = Tt - 1 - x;
  } else {
    double d = 3.0 * (double)(x - a) * (double)(x - a) / (double)a + (double)a;
    v = Tt - 1 - (int)ceil(d);
  }
  keep[i] = v;
}

// ---- kernel: q/k/v projections (bf16 WMMA, f32 accumulate) --------------
// grid.y: 0 = q (gathered rows), 1 = k (row-major), 2 = v (stored transposed)

__global__ void proj_kernel(const __bf16* __restrict__ xb,
                            const __bf16* __restrict__ Wb,   // [3][64][1024] bf16
                            const float* __restrict__ bq,
                            const float* __restrict__ bk,
                            const float* __restrict__ bv,
                            const int* __restrict__ keep,
                            __bf16* __restrict__ qg,
                            __bf16* __restrict__ kb,
                            __bf16* __restrict__ vT) {
  const int lane  = threadIdx.x & 31;
  const int w     = threadIdx.x >> 5;
  const int which = blockIdx.y;
  const int tile  = blockIdx.x * 4 + w;
  const int nrows = (which == 0) ? (Bb * KQ) : (Bb * Tt);
  if (tile * 16 >= nrows) return;

  const int m    = lane & 15;
  const int half = lane >> 4;

  const __bf16* W    = Wb + (size_t)which * (HSs * Cc);
  const float*  bias = (which == 0) ? bq : (which == 1) ? bk : bv;

  size_t xrow;
  {
    int g = tile * 16 + m;
    if (which == 0) {
      int bbi = g / KQ, ii = g - bbi * KQ;
      xrow = (size_t)bbi * Tt + keep[ii];
    } else {
      xrow = (size_t)g;
    }
  }
  const __bf16* xr = xb + xrow * Cc;

  const __bf16* wr[4];
#pragma unroll
  for (int n = 0; n < 4; ++n) wr[n] = W + (size_t)(n * 16 + m) * Cc;

  v8f acc[4];
#pragma unroll
  for (int n = 0; n < 4; ++n) acc[n] = {};

  for (int k0 = 0; k0 < Cc; k0 += 32) {
    const v16bf a = pack16(*(const v8bf*)(xr + k0 + half * 8),
                           *(const v8bf*)(xr + k0 + 16 + half * 8));
#pragma unroll
    for (int n = 0; n < 4; ++n) {
      const v16bf bw = pack16(*(const v8bf*)(wr[n] + k0 + half * 8),
                              *(const v8bf*)(wr[n] + k0 + 16 + half * 8));
      acc[n] = __builtin_amdgcn_wmma_f32_16x16x32_bf16(
          false, a, false, bw, (short)0, acc[n], false, false);
    }
  }

#pragma unroll
  for (int n = 0; n < 4; ++n) {
    const int ncol = n * 16 + m;
    const float bb = bias[ncol];
#pragma unroll
    for (int r = 0; r < 8; ++r) {
      const int mr  = r + 8 * half;
      const int g   = tile * 16 + mr;
      const float v = acc[n][r] + bb;
      if (which == 2) {
        int bbi = g >> 12;              // / Tt
        int t   = g & (Tt - 1);
        vT[((size_t)bbi * HSs + ncol) * Tt + t] = f2bf(v);
      } else if (which == 1) {
        kb[(size_t)g * HSs + ncol] = f2bf(v);
      } else {
        qg[(size_t)g * HSs + ncol] = f2bf(v);
      }
    }
  }
}

// ---- kernel: masked flash attention over kept queries -------------------

#if HAVE_ASYNC
__device__ __forceinline__ void stage_k(__bf16* dstLds, const __bf16* src, int lane) {
#pragma unroll
  for (int c = 0; c < 8; ++c) {
    const int idx = c * 32 + lane;     // 16-byte chunk index (32 keys x 128B)
    __builtin_amdgcn_global_load_async_to_lds_b128(
        (gv4i*)(src + idx * 8),
        (sv4i*)(dstLds + idx * 8),
        0, 0);
  }
}
#endif

__global__ void attn_kernel(const __bf16* __restrict__ qg,
                            const __bf16* __restrict__ kb,
                            const __bf16* __restrict__ vT,
                            const int* __restrict__ keep,
                            float* __restrict__ out) {
  __shared__ __align__(16) __bf16 plds[4][16][32];
#if HAVE_ASYNC
  __shared__ __align__(16) __bf16 kstage[4][2][32][HSs];   // 32 KB
#endif

  const int lane = threadIdx.x & 31;
  const int w    = threadIdx.x >> 5;
  const int qt   = blockIdx.x * 4 + w;
  const int b    = blockIdx.y;
  const int col  = lane & 15;
  const int half = lane >> 4;
  const float scale = 0.03125f;        // C^-0.5 = 1/32

  // Q fragments held for the whole kernel (A layout: lane = M row)
  const __bf16* qr = qg + ((size_t)b * KQ + qt * 16 + col) * HSs;
  const v16bf aq0 = pack16(*(const v8bf*)(qr + half * 8),
                           *(const v8bf*)(qr + 16 + half * 8));
  const v16bf aq1 = pack16(*(const v8bf*)(qr + 32 + half * 8),
                           *(const v8bf*)(qr + 48 + half * 8));

  int krow[8];
#pragma unroll
  for (int r = 0; r < 8; ++r) krow[r] = keep[qt * 16 + r + 8 * half];
  const int kmax = keep[qt * 16 + 15];

  float rm[8], rs[8];
#pragma unroll
  for (int r = 0; r < 8; ++r) { rm[r] = -3.0e38f; rs[r] = 0.0f; }
  v8f o[4];
#pragma unroll
  for (int n = 0; n < 4; ++n) o[n] = {};

#if HAVE_ASYNC
  stage_k(&kstage[w][0][0][0], kb + (size_t)b * Tt * HSs, lane);
#endif

  for (int t0 = 0; t0 <= kmax; t0 += 32) {
#if HAVE_ASYNC
    const int cur = (t0 >> 5) & 1;
    if (t0 + 32 <= kmax) {
      stage_k(&kstage[w][cur ^ 1][0][0],
              kb + ((size_t)b * Tt + t0 + 32) * HSs, lane);
      WAIT_ASYNC(8);                  // current tile's copies are done
    } else {
      WAIT_ASYNC(0);
    }
    asm volatile("" ::: "memory");
    const __bf16* kr0 = &kstage[w][cur][col][0];
    const __bf16* kr1 = &kstage[w][cur][16 + col][0];
#else
    const __bf16* kr0 = kb + ((size_t)b * Tt + t0 + col) * HSs;
    const __bf16* kr1 = kr0 + 16 * HSs;
    __builtin_prefetch(kr0 + 32 * HSs, 0, 1);
#endif

    // ---- S = qg . k^T for 32 keys (two 16x16 tiles) ----
    const v16bf bk00 = pack16(*(const v8bf*)(kr0 + half * 8),
                              *(const v8bf*)(kr0 + 16 + half * 8));
    const v16bf bk01 = pack16(*(const v8bf*)(kr0 + 32 + half * 8),
                              *(const v8bf*)(kr0 + 48 + half * 8));
    const v16bf bk10 = pack16(*(const v8bf*)(kr1 + half * 8),
                              *(const v8bf*)(kr1 + 16 + half * 8));
    const v16bf bk11 = pack16(*(const v8bf*)(kr1 + 32 + half * 8),
                              *(const v8bf*)(kr1 + 48 + half * 8));
    const v8f zero = {};
    v8f s0 = __builtin_amdgcn_wmma_f32_16x16x32_bf16(false, aq0, false, bk00, (short)0, zero, false, false);
    s0     = __builtin_amdgcn_wmma_f32_16x16x32_bf16(false, aq1, false, bk01, (short)0, s0,   false, false);
    v8f s1 = __builtin_amdgcn_wmma_f32_16x16x32_bf16(false, aq0, false, bk10, (short)0, zero, false, false);
    s1     = __builtin_amdgcn_wmma_f32_16x16x32_bf16(false, aq1, false, bk11, (short)0, s1,   false, false);

    // ---- mask + online softmax (row stats across 16-lane halves) ----
    float p0a[8], p1a[8], corr[8];
#pragma unroll
    for (int r = 0; r < 8; ++r) {
      float a0 = s0[r] * scale;
      float a1 = s1[r] * scale;
      if (t0 + col > krow[r])      a0 = -3.0e38f;
      if (t0 + 16 + col > krow[r]) a1 = -3.0e38f;
      float tm = rmax16(fmaxf(a0, a1));
      float mn = fmaxf(rm[r], tm);
      corr[r]  = __expf(rm[r] - mn);
      rm[r]    = mn;
      p0a[r]   = __expf(a0 - mn);
      p1a[r]   = __expf(a1 - mn);
      rs[r]    = rs[r] * corr[r] + rsum16(p0a[r] + p1a[r]);
    }

#pragma unroll
    for (int n = 0; n < 4; ++n)
#pragma unroll
      for (int r = 0; r < 8; ++r) o[n][r] *= corr[r];

    // ---- re-layout P (C/D format) -> A format via wave-private LDS ----
#pragma unroll
    for (int r = 0; r < 8; ++r) {
      plds[w][r + 8 * half][col]      = f2bf(p0a[r]);
      plds[w][r + 8 * half][16 + col] = f2bf(p1a[r]);
    }
    asm volatile("" ::: "memory");   // DS ops are in-order within a wave
    const v16bf pf = pack16(*(const v8bf*)&plds[w][col][half * 8],
                            *(const v8bf*)&plds[w][col][16 + half * 8]);

    // ---- O += P . V (vT is HS-major so B-frags load contiguously) ----
#pragma unroll
    for (int n = 0; n < 4; ++n) {
      const __bf16* vr = vT + ((size_t)b * HSs + n * 16 + col) * Tt + t0;
      __builtin_prefetch(vr + 32, 0, 1);
      const v16bf bvf = pack16(*(const v8bf*)(vr + half * 8),
                               *(const v8bf*)(vr + 16 + half * 8));
      o[n] = __builtin_amdgcn_wmma_f32_16x16x32_bf16(
          false, pf, false, bvf, (short)0, o[n], false, false);
    }
  }

  float inv[8];
#pragma unroll
  for (int r = 0; r < 8; ++r) inv[r] = 1.0f / rs[r];
#pragma unroll
  for (int n = 0; n < 4; ++n)
#pragma unroll
    for (int r = 0; r < 8; ++r) {
      const int mr = r + 8 * half;
      out[((size_t)b * KQ + qt * 16 + mr) * HSs + n * 16 + col] = o[n][r] * inv[r];
    }
}

// ---- host launcher -------------------------------------------------------

extern "C" void kernel_launch(void* const* d_in, const int* in_sizes, int n_in,
                              void* d_out, int out_size, void* d_ws, size_t ws_size,
                              hipStream_t stream) {
  const float* x  = (const float*)d_in[0];
  const float* Wq = (const float*)d_in[1];
  const float* bq = (const float*)d_in[2];
  const float* Wk = (const float*)d_in[3];
  const float* bk = (const float*)d_in[4];
  const float* Wv = (const float*)d_in[5];
  const float* bv = (const float*)d_in[6];
  float* out = (float*)d_out;

  char* ws = (char*)d_ws;
  int*    keep = (int*)ws;                               // 8 KB
  __bf16* qg   = (__bf16*)(ws + (1 << 16));              // 2 MB (B,KQ,HS)
  __bf16* kbf  = qg  + (size_t)Bb * KQ * HSs;            // 4 MB (B,T,HS)
  __bf16* vT   = kbf + (size_t)Bb * Tt * HSs;            // 4 MB (B,HS,T)
  __bf16* Wb   = vT  + (size_t)Bb * Tt * HSs;            // 384 KB [3][64][1024]
  __bf16* xb   = Wb  + (size_t)3 * HSs * Cc;             // 64 MB (B,T,C)

  const long long nx = (long long)Bb * Tt * Cc;
  const long long nw = (long long)HSs * Cc;

  cvt_kernel<<<dim3((unsigned)((nx / 8 + 255) / 256)), dim3(256), 0, stream>>>(x, xb, nx);
  cvt_kernel<<<dim3((unsigned)((nw / 8 + 255) / 256)), dim3(256), 0, stream>>>(Wq, Wb, nw);
  cvt_kernel<<<dim3((unsigned)((nw / 8 + 255) / 256)), dim3(256), 0, stream>>>(Wk, Wb + nw, nw);
  cvt_kernel<<<dim3((unsigned)((nw / 8 + 255) / 256)), dim3(256), 0, stream>>>(Wv, Wb + 2 * nw, nw);
  keep_kernel<<<dim3((KQ + 255) / 256), dim3(256), 0, stream>>>(keep);
  proj_kernel<<<dim3((Bb * Tt / 16) / 4, 3), dim3(128), 0, stream>>>(
      xb, Wb, bq, bk, bv, keep, qg, kbf, vT);
  attn_kernel<<<dim3(KQ / 16 / 4, Bb), dim3(128), 0, stream>>>(qg, kbf, vT, keep, out);
}
